// MSARowAttentionWithPairBias_71536975282374
// MI455X (gfx1250) — matrix operand order repaired and checked by reading, not compiled, not run on hardware
//
#include <hip/hip_runtime.h>

// ---------------------------------------------------------------------------
// MSA row attention with pair bias, MI455X (gfx1250, wave32, WMMA bf16).
//  * Fused per-s attention in 312 KB LDS (CDNA5 allows 320 KB/WG): pair-bias
//    GEMM + q.q^T + softmax-over-query-axis + a@q never touch HBM.
//  * Fused layernorm + q/g projection (LDS bf16 panel, no x_bf round-trip).
//  * out_proj stages its A panel in LDS: cat read once, not 16x.
//  * LDS staging uses CDNA5 async global->LDS loads (ASYNCcnt).
// ---------------------------------------------------------------------------

typedef __attribute__((ext_vector_type(16))) __bf16 v16bf;
typedef __attribute__((ext_vector_type(8)))  float  v8f;

#define S_  128
#define R_  384
#define CM_ 256
#define CZ_ 128
#define C_  32
#define H_  8
#define RR_ (R_ * R_)      // 147456
#define SR_ (S_ * R_)      // 49152

// LDS layout for the fused attention kernel (312 KB total).
#define SC_BYTES   (R_ * 128 * 4)   // 196608: f32 scores, 384 x 128 j-slab
#define X_BYTES    (R_ * 128 * 2)   //  98304: bias-A stage -> bf16 attn wts
#define Q_BYTES    (R_ * C_ * 2)    //  24576: q[s] 384x32 bf16
#define SMEM_BYTES (SC_BYTES + X_BYTES + Q_BYTES)  // 319488 <= 327680

__device__ __forceinline__ v8f wmma_bf16(v16bf a, v16bf b, v8f c) {
  return __builtin_amdgcn_wmma_f32_16x16x32_bf16(false, a, false, b, (short)0, c,
                                                 false, false);
}

// Async 16-byte global -> LDS copy (CDNA5, tracked by ASYNCcnt).
__device__ __forceinline__ void async_copy16(unsigned lds_off,
                                             unsigned long long gaddr) {
  asm volatile("global_load_async_to_lds_b128 %0, %1, off"
               :: "v"(lds_off), "v"(gaddr) : "memory");
}
__device__ __forceinline__ void wait_async0() {
  asm volatile("s_wait_asynccnt 0" ::: "memory");
}

// A-matrix 16x32 bf16 tile: lanes 0-15 M=lane, K=0..15;
// lanes 16-31 M=lane-16, K=16..31; VGPR v packs (k,k+1).
__device__ __forceinline__ v16bf load_a_tile(const __bf16* __restrict__ src,
                                             int ld, long m0, int k0, int lane) {
  const int half = lane >> 4, m = lane & 15;
  const __bf16* row = src + (size_t)(m0 + m) * ld + k0;
  v16bf a;
#pragma unroll
  for (int v = 0; v < 8; ++v) {
    const int kb = 16 * half + 2 * v;
    a[2 * v]     = row[kb];
    a[2 * v + 1] = row[kb + 1];
  }
  return a;
}

// B-matrix 32x16 (KxN) tile, source row-major (N x K): B[k][n] = W[n*ld+k].
__device__ __forceinline__ v16bf load_b_nk(const __bf16* __restrict__ W,
                                           int ld, int k0, int n0, int lane) {
  const int half = lane >> 4, n = lane & 15;
  const __bf16* row = W + (size_t)(n0 + n) * ld + k0 + 16 * half;
  v16bf b;
#pragma unroll
  for (int v = 0; v < 8; ++v) {
    b[2 * v]     = row[2 * v];
    b[2 * v + 1] = row[2 * v + 1];
  }
  return b;
}

// B-matrix 32x16 (KxN) tile, source row-major (K x N): B[k][n] = W[k*ld+n].
__device__ __forceinline__ v16bf load_b_kn(const __bf16* __restrict__ W,
                                           int ld, int k0, int n0, int lane) {
  const int half = lane >> 4, n = lane & 15;
  v16bf b;
#pragma unroll
  for (int v = 0; v < 8; ++v) {
    const int k = k0 + 16 * half + 2 * v;
    b[2 * v]     = W[(size_t)k * ld + n0 + n];
    b[2 * v + 1] = W[(size_t)(k + 1) * ld + n0 + n];
  }
  return b;
}

// ---------------------------------------------------------------------------
// Pair layernorm (stats only; scale/shift folded into per-head W'/beta).
__global__ __launch_bounds__(256) void pair_ln_kernel(
    const float* __restrict__ pair, __bf16* __restrict__ pn) {
  const int lane = threadIdx.x & 31, wid = threadIdx.x >> 5;
  const size_t row = (size_t)blockIdx.x * 8 + wid;
  const float* r = pair + row * CZ_;
  float v[4], s = 0.f;
#pragma unroll
  for (int k = 0; k < 4; ++k) { v[k] = r[lane + 32 * k]; s += v[k]; }
#pragma unroll
  for (int m = 16; m; m >>= 1) s += __shfl_xor(s, m, 32);
  const float mu = s * (1.f / CZ_);
  float q = 0.f;
#pragma unroll
  for (int k = 0; k < 4; ++k) { const float d = v[k] - mu; q += d * d; }
#pragma unroll
  for (int m = 16; m; m >>= 1) q += __shfl_xor(q, m, 32);
  const float rstd = rsqrtf(q * (1.f / CZ_) + 1e-5f);
  __bf16* o = pn + row * CZ_;
#pragma unroll
  for (int k = 0; k < 4; ++k) o[lane + 32 * k] = (__bf16)((v[k] - mu) * rstd);
}

__global__ void conv_bf16_kernel(const float* __restrict__ src,
                                 __bf16* __restrict__ dst, int n) {
  for (int i = blockIdx.x * blockDim.x + threadIdx.x; i < n;
       i += gridDim.x * blockDim.x)
    dst[i] = (__bf16)src[i];
}

// W'[h,t,c] = bias_w * ln_pair_w (bf16), beta[h,t] = sum_c ln_pair_b * bias_w.
__global__ __launch_bounds__(128) void wp_beta_kernel(
    const float* __restrict__ bias_w, const float* __restrict__ lnw,
    const float* __restrict__ lnb, __bf16* __restrict__ Wp,
    float* __restrict__ beta) {
  const int blk = blockIdx.x;          // h * S_ + t
  const int h = blk >> 7;              // S_ == 128
  const int c = threadIdx.x;
  const float bw = bias_w[(size_t)blk * CZ_ + c];
  Wp[(size_t)blk * CZ_ + c] = (__bf16)(bw * lnw[h * CZ_ + c]);
  float part = bw * lnb[h * CZ_ + c];
#pragma unroll
  for (int m = 16; m; m >>= 1) part += __shfl_xor(part, m, 32);
  __shared__ float red[4];
  if ((threadIdx.x & 31) == 0) red[threadIdx.x >> 5] = part;
  __syncthreads();
  if (threadIdx.x == 0) beta[blk] = red[0] + red[1] + red[2] + red[3];
}

// ---------------------------------------------------------------------------
// Fused cumulative layernorm + q/g projection.
// Block = 32 msa rows: LN them into a 16 KB bf16 LDS panel (also writing the
// f32 x for the next head), then 8 waves run {2 mtiles} x {q n0/n1, g n0/n1}
// WMMA jobs from LDS. No x_bf16 global round-trip.
__global__ __launch_bounds__(256) void ln_qg_kernel(
    const float* __restrict__ src, float* __restrict__ xout,
    const float* __restrict__ lnw, const float* __restrict__ lnb,
    const __bf16* __restrict__ qw, const __bf16* __restrict__ gw,
    __bf16* __restrict__ qbf, float* __restrict__ gout) {
  __shared__ __bf16 xa[32 * CM_];                 // 16 KB
  const int lane = threadIdx.x & 31, wave = threadIdx.x >> 5;
  const size_t row0 = (size_t)blockIdx.x * 32;

  // LN: each wave handles 4 consecutive rows.
#pragma unroll
  for (int rr = 0; rr < 4; ++rr) {
    const int lrow = wave * 4 + rr;
    const float* r = src + (row0 + lrow) * CM_;
    float v[8], s = 0.f;
#pragma unroll
    for (int k = 0; k < 8; ++k) { v[k] = r[lane + 32 * k]; s += v[k]; }
#pragma unroll
    for (int m = 16; m; m >>= 1) s += __shfl_xor(s, m, 32);
    const float mu = s * (1.f / CM_);
    float q = 0.f;
#pragma unroll
    for (int k = 0; k < 8; ++k) { const float d = v[k] - mu; q += d * d; }
#pragma unroll
    for (int m = 16; m; m >>= 1) q += __shfl_xor(q, m, 32);
    const float rstd = rsqrtf(q * (1.f / CM_) + 1e-5f);
#pragma unroll
    for (int k = 0; k < 8; ++k) {
      const int c = lane + 32 * k;
      const float y = (v[k] - mu) * rstd * lnw[c] + lnb[c];
      xout[(row0 + lrow) * CM_ + c] = y;
      xa[lrow * CM_ + c]            = (__bf16)y;
    }
  }
  __syncthreads();

  // GEMM from LDS panel. Wave-uniform job select keeps EXEC all-ones.
  const int mt  = wave >> 2;                      // 0 or 1
  const int job = wave & 3;                       // 0,1: q ; 2,3: g
  const int n0  = (job & 1) * 16;
  const __bf16* W = (job < 2) ? qw : gw;
  v8f acc = {};
#pragma unroll
  for (int kt = 0; kt < CM_ / 32; ++kt) {
    const v16bf a = load_a_tile(xa + mt * 16 * CM_, CM_, 0, kt * 32, lane);
    const v16bf b = load_b_nk(W, CM_, kt * 32, n0, lane);
    acc = wmma_bf16(a, b, acc);
  }
  const int half = lane >> 4, n = n0 + (lane & 15);
  if (job < 2) {
#pragma unroll
    for (int r = 0; r < 8; ++r)
      qbf[(row0 + mt * 16 + r + 8 * half) * C_ + n] = (__bf16)acc[r];
  } else {
#pragma unroll
    for (int r = 0; r < 8; ++r)
      gout[(row0 + mt * 16 + r + 8 * half) * C_ + n] =
          1.f / (1.f + __expf(-acc[r]));
  }
}

// ---------------------------------------------------------------------------
// Fused per-s attention: bias GEMM + q.q^T scores + softmax over the QUERY
// axis i + a@q, all in LDS. One workgroup (256 threads = 8 waves) per s.
// Slab loop over j in 3 chunks of 128; AV accumulators persist across slabs.
__global__ __launch_bounds__(256) void attn_fused_kernel(
    const __bf16* __restrict__ qbf, const __bf16* __restrict__ pn,
    const __bf16* __restrict__ Wp, const float* __restrict__ beta_h,
    const float* __restrict__ g, __bf16* __restrict__ catbf, int head) {
  extern __shared__ char smem[];
  float*  sc   = (float*)smem;                          // [384][128] f32
  __bf16* xreg = (__bf16*)(smem + SC_BYTES);            // biasA then attn wts
  __bf16* qlds = (__bf16*)(smem + SC_BYTES + X_BYTES);  // [384][32] bf16

  const int s = blockIdx.x;
  const int lane = threadIdx.x & 31, wave = threadIdx.x >> 5;
  const float scale = 0.17677669529663687f;             // 1/sqrt(32)

  // Stage q[s] (24 KB) into LDS asynchronously.
  {
    const unsigned long long qsrc =
        (unsigned long long)(uintptr_t)(qbf + (size_t)s * R_ * C_);
    const unsigned qbase = (unsigned)(uintptr_t)qlds;
#pragma unroll
    for (int it = 0; it < 6; ++it) {
      const unsigned off = (threadIdx.x + 256u * it) * 16u;
      async_copy16(qbase + off, qsrc + off);
    }
  }

  // Per-wave invariants: score n-tile n0s = wave*16, AV column tile n0c.
  const int n0s = wave * 16;
  const int n0c = (wave & 1) * 16;

  // W' B-tiles for the bias GEMM (global, hot in L2; invariant per wave).
  v16bf bw[4];
#pragma unroll
  for (int kt = 0; kt < 4; ++kt) bw[kt] = load_b_nk(Wp, CZ_, kt * 32, n0s, lane);
  const float bet = beta_h[n0s + (lane & 15)];

  v8f oacc[6];
#pragma unroll
  for (int t = 0; t < 6; ++t) oacc[t] = (v8f){};

  const unsigned xbase = (unsigned)(uintptr_t)xreg;
  const int half = lane >> 4;

  for (int jb = 0; jb < 3; ++jb) {
    // ---- stage bias-A slab: pair_norm rows 1152 s + 3 i + jb, i = 0..383 --
    {
      const unsigned long long pbase = (unsigned long long)(uintptr_t)(
          pn + ((size_t)1152 * s + jb) * CZ_);
#pragma unroll
      for (int it = 0; it < 24; ++it) {
        const int id = threadIdx.x + 256 * it;   // 0..6143
        const int rowi = id >> 4, part = id & 15;
        async_copy16(xbase + rowi * 256u + part * 16u,
                     pbase + (size_t)rowi * 768u + part * 16u);
      }
      wait_async0();
      __syncthreads();
    }
    const __bf16* biasA = xreg;

    // ---- scores: wave handles n-tile n0s for all 24 m-tiles -------------
    const v16bf bq = load_b_nk(qlds, C_, 0, jb * 128 + n0s, lane);
    for (int t = 0; t < 24; ++t) {
      const int m0 = t * 16;
      v8f acc = {};
#pragma unroll
      for (int kt = 0; kt < 4; ++kt)
        acc = wmma_bf16(load_a_tile(biasA, CZ_, m0, kt * 32, lane), bw[kt], acc);
      acc = wmma_bf16(load_a_tile(qlds, C_, m0, 0, lane), bq, acc);
      const int n = n0s + (lane & 15);
#pragma unroll
      for (int r = 0; r < 8; ++r)
        sc[(m0 + r + 8 * half) * 128 + n] = (acc[r] + bet) * scale;
    }
    __syncthreads();

    // ---- softmax over i (query axis) for each column jl of the slab -----
    __bf16* al = xreg;   // overwrite the bias-A region with bf16 attn weights
    if (threadIdx.x < 128) {
      const int jl = threadIdx.x;
      float m = -3.0e38f, d = 0.f;
      for (int i = 0; i < R_; ++i) {
        const float v = sc[i * 128 + jl];
        const float nm = fmaxf(m, v);
        d = d * __expf(m - nm) + __expf(v - nm);
        m = nm;
      }
      const float inv = 1.f / d;
      for (int i = 0; i < R_; ++i)
        al[i * 128 + jl] = (__bf16)(__expf(sc[i * 128 + jl] - m) * inv);
    }
    __syncthreads();

    // ---- AV partial: o[i,c] += sum_{j in slab} a[i,j] q[j,c] ------------
    v16bf bv[4];
#pragma unroll
    for (int kt = 0; kt < 4; ++kt)
      bv[kt] = load_b_kn(qlds, C_, jb * 128 + kt * 32, n0c, lane);
#pragma unroll
    for (int t = 0; t < 6; ++t) {
      const int m0 = (((wave + 8 * t) >> 1)) * 16;
#pragma unroll
      for (int kt = 0; kt < 4; ++kt)
        oacc[t] = wmma_bf16(load_a_tile(al, 128, m0, kt * 32, lane), bv[kt],
                            oacc[t]);
    }
    __syncthreads();   // before the next slab overwrites sc / xreg
  }

  // ---- epilogue: o *= g, write bf16 concat channel slice ----------------
#pragma unroll
  for (int t = 0; t < 6; ++t) {
    const int ot = wave + 8 * t;
    const int m0 = (ot >> 1) * 16;
    const int c = n0c + (lane & 15);
#pragma unroll
    for (int r = 0; r < 8; ++r) {
      const size_t row = (size_t)s * R_ + m0 + r + 8 * half;
      catbf[row * CM_ + head * C_ + c] =
          (__bf16)(g[row * C_ + c] * oacc[t][r]);
    }
  }
}

// out = concat @ out_w^T + out_b  ->  f32 d_out.
// Block = one 16-row M panel staged in LDS (async); 8 waves x 2 n-tiles.
__global__ __launch_bounds__(256) void out_proj_kernel(
    const __bf16* __restrict__ catbf, const __bf16* __restrict__ ow,
    const float* __restrict__ ob, float* __restrict__ out) {
  __shared__ __bf16 xa[16 * CM_];                 // 8 KB
  const int lane = threadIdx.x & 31, wave = threadIdx.x >> 5;
  const size_t m0 = (size_t)blockIdx.x * 16;

  {
    const unsigned base = (unsigned)(uintptr_t)xa;
    const unsigned long long gsrc =
        (unsigned long long)(uintptr_t)(catbf + m0 * CM_);
#pragma unroll
    for (int it = 0; it < 2; ++it) {
      const unsigned off = (threadIdx.x + 256u * it) * 16u;
      async_copy16(base + off, gsrc + off);
    }
    wait_async0();
    __syncthreads();
  }

  const int n0 = wave * 16;
  v8f acc0 = {}, acc1 = {};
#pragma unroll
  for (int kt = 0; kt < CM_ / 32; ++kt) {
    const v16bf a  = load_a_tile(xa, CM_, 0, kt * 32, lane);
    const v16bf b0 = load_b_nk(ow, CM_, kt * 32, n0, lane);
    const v16bf b1 = load_b_nk(ow, CM_, kt * 32, n0 + 128, lane);
    acc0 = wmma_bf16(a, b0, acc0);
    acc1 = wmma_bf16(a, b1, acc1);
  }
  const int half = lane >> 4, n = n0 + (lane & 15);
  const float bias0 = ob[n], bias1 = ob[n + 128];
#pragma unroll
  for (int r = 0; r < 8; ++r) {
    out[(m0 + r + 8 * half) * CM_ + n]       = acc0[r] + bias0;
    out[(m0 + r + 8 * half) * CM_ + n + 128] = acc1[r] + bias1;
  }
}

// ---------------------------------------------------------------------------
extern "C" void kernel_launch(void* const* d_in, const int* in_sizes, int n_in,
                              void* d_out, int out_size, void* d_ws,
                              size_t ws_size, hipStream_t stream) {
  (void)in_sizes; (void)n_in; (void)out_size; (void)ws_size;
  const float* msa       = (const float*)d_in[0];   // (S,R,CM)
  const float* pair      = (const float*)d_in[1];   // (R,R,CZ)
  const float* ln_msa_w  = (const float*)d_in[2];   // (H,CM)
  const float* ln_msa_b  = (const float*)d_in[3];   // (H,CM)
  const float* qkv_w     = (const float*)d_in[4];   // (H,C,CM)
  const float* ln_pair_w = (const float*)d_in[5];   // (H,CZ)
  const float* ln_pair_b = (const float*)d_in[6];   // (H,CZ)
  const float* bias_w    = (const float*)d_in[7];   // (H,S,CZ)
  const float* g_w       = (const float*)d_in[8];   // (H,C,CM)
  const float* out_w     = (const float*)d_in[9];   // (CM, C*H)
  const float* out_b     = (const float*)d_in[10];  // (CM)
  float* out = (float*)d_out;

  // Workspace carve (~125 MB total).
  char* p = (char*)d_ws;
  auto carve = [&](size_t bytes) -> void* {
    void* r = (void*)p;
    p += (bytes + 255) & ~(size_t)255;
    return r;
  };
  float*  x       = (float*) carve((size_t)SR_ * CM_ * 4);   // 50.3 MB
  __bf16* pair_n  = (__bf16*)carve((size_t)RR_ * CZ_ * 2);   // 37.7 MB
  __bf16* Wp      = (__bf16*)carve((size_t)H_ * S_ * CZ_ * 2);
  float*  beta    = (float*) carve((size_t)H_ * S_ * 4);
  __bf16* qkvw_bf = (__bf16*)carve((size_t)H_ * C_ * CM_ * 2);
  __bf16* gw_bf   = (__bf16*)carve((size_t)H_ * C_ * CM_ * 2);
  __bf16* outw_bf = (__bf16*)carve((size_t)CM_ * C_ * H_ * 2);
  __bf16* q_bf    = (__bf16*)carve((size_t)SR_ * C_ * 2);    // 3.1 MB
  float*  g       = (float*) carve((size_t)SR_ * C_ * 4);    // 6.3 MB
  __bf16* cat_bf  = (__bf16*)carve((size_t)SR_ * CM_ * 2);   // 25.2 MB

  // Allow 312 KB dynamic LDS for the fused attention kernel (CDNA5 WGP).
  (void)hipFuncSetAttribute(reinterpret_cast<const void*>(attn_fused_kernel),
                            hipFuncAttributeMaxDynamicSharedMemorySize,
                            SMEM_BYTES);

  // Head-independent prep.
  pair_ln_kernel<<<RR_ / 8, 256, 0, stream>>>(pair, pair_n);
  conv_bf16_kernel<<<256, 256, 0, stream>>>(qkv_w, qkvw_bf, H_ * C_ * CM_);
  conv_bf16_kernel<<<256, 256, 0, stream>>>(g_w, gw_bf, H_ * C_ * CM_);
  conv_bf16_kernel<<<256, 256, 0, stream>>>(out_w, outw_bf, CM_ * C_ * H_);
  wp_beta_kernel<<<H_ * S_, 128, 0, stream>>>(bias_w, ln_pair_w, ln_pair_b, Wp,
                                              beta);

  for (int h = 0; h < H_; ++h) {
    ln_qg_kernel<<<SR_ / 32, 256, 0, stream>>>(
        h == 0 ? msa : x, x, ln_msa_w + h * CM_, ln_msa_b + h * CM_,
        qkvw_bf + (size_t)h * C_ * CM_, gw_bf + (size_t)h * C_ * CM_, q_bf, g);
    attn_fused_kernel<<<S_, 256, SMEM_BYTES, stream>>>(
        q_bf, pair_n, Wp + (size_t)h * S_ * CZ_, beta + h * S_, g, cat_bf, h);
  }
  out_proj_kernel<<<SR_ / 16, 256, 0, stream>>>(cat_bf, outw_bf, out_b, out);
}
